// LinearAndCEL_20383914787522
// MI455X (gfx1250) — matrix operand order repaired and checked by reading, not compiled, not run on hardware
//
#include <hip/hip_runtime.h>

// Problem constants (from the reference): B=8, S=1024 -> N=8192 rows, C=2048, V=50257
#define NROW 8192
#define CDIM 2048
#define VDIM 50257
#define VTILE 256                               // vocab columns per workgroup (8 waves x 32)
#define NVT ((VDIM + VTILE - 1) / VTILE)        // 197 vocab tiles

typedef __attribute__((ext_vector_type(16))) __bf16 v16bf;
typedef __attribute__((ext_vector_type(8)))  __bf16 v8bf;
typedef __attribute__((ext_vector_type(4)))  __bf16 v4bf;
typedef __attribute__((ext_vector_type(8)))  float  v8f;

union Frag32B { v16bf v; uint4 q[2]; };

// B fragment (W^T, 32x16): lane holds K chunks {k+8h..+7, k+16+8h..+7} of one W row.
__device__ __forceinline__ void loadB(Frag32B& f, const __bf16* __restrict__ p, int k) {
  f.q[0] = *(const uint4*)(p + k);
  f.q[1] = *(const uint4*)(p + k + 16);
}
// A fragment (x, 16x32): lane holds K chunks {k+8h..+7, k+16+8h..+7}.
__device__ __forceinline__ void loadA(Frag32B& f, const __bf16* __restrict__ p, int k) {
  f.q[0] = *(const uint4*)(p + k);
  f.q[1] = *(const uint4*)(p + k + 16);
}

#define WMMA_BF16(A, B, C) \
  __builtin_amdgcn_wmma_f32_16x16x32_bf16(false, (A), false, (B), (short)0, (C), false, false)

// ---------------------------------------------------------------------------
// Kernel 1/2: fp32 -> bf16 conversion (x and W staged once into workspace).
// ---------------------------------------------------------------------------
__global__ __launch_bounds__(256) void cvt_f32_bf16(const float* __restrict__ in,
                                                    __bf16* __restrict__ out, long n4) {
  long i = (long)blockIdx.x * 256 + threadIdx.x;
  if (i >= n4) return;
  float4 f = ((const float4*)in)[i];
  v4bf o;
  o[0] = (__bf16)f.x; o[1] = (__bf16)f.y; o[2] = (__bf16)f.z; o[3] = (__bf16)f.w;
  ((v4bf*)out)[i] = o;
}

// ---------------------------------------------------------------------------
// Kernel 3: fused GEMM tile + stable per-tile logsumexp partials.
// Grid: (NROW/64, NVT). Block: 256 threads = 8 waves.
// Each wave: 32 vocab cols (2 B-frags) x 64 rows (4 A-frags) -> 8 WMMAs per
// 12 loads per 32-K chunk, with an explicit 2-deep ping-pong pipeline so the
// WMMA bursts only wait on loads issued one phase earlier.
// ---------------------------------------------------------------------------
__global__ __launch_bounds__(256) void gemm_lse_partials(
    const __bf16* __restrict__ xb, const __bf16* __restrict__ wb,
    const float* __restrict__ bias, float2* __restrict__ partials) {
  __shared__ float lm[8][64];
  __shared__ float lz[8][64];

  const int wave = threadIdx.x >> 5;
  const int lane = threadIdx.x & 31;
  const int half = lane >> 4;      // 0: lanes 0-15, 1: lanes 16-31
  const int l16  = lane & 15;
  const int row0 = blockIdx.x * 64;
  const int colbase = blockIdx.y * VTILE + wave * 32;
  const int col0 = colbase + l16;
  const int col1 = colbase + 16 + l16;
  const bool cv0 = col0 < VDIM;
  const bool cv1 = col1 < VDIM;
  const long cc0 = cv0 ? col0 : 0;           // clamp OOB vocab loads
  const long cc1 = cv1 ? col1 : 0;

  const __bf16* pB0 = wb + cc0 * (long)CDIM + half * 8;
  const __bf16* pB1 = wb + cc1 * (long)CDIM + half * 8;
  const __bf16* pA  = xb + (long)(row0 + l16) * CDIM + half * 8;

  v8f acc[2][4] = {};  // [col-subtile][row-subtile] f32 accumulators (64 VGPRs)

  // Ping-pong fragment buffers; phase 0 preloaded, then alternate.
  Frag32B b0[2], b1[2], a[2][4];
  loadB(b0[0], pB0, 0);
  loadB(b1[0], pB1, 0);
#pragma unroll
  for (int m = 0; m < 4; ++m) loadA(a[0][m], pA + (long)m * 16 * CDIM, 0);

  // NOTE: the final phase issues loads for k==CDIM; those overread <=64B past a
  // row and land in the next workspace region (xb->wb, wb->partials) - harmless.
  for (int k = 0; k < CDIM; k += 64) {
    // Issue loads for k+32 into buffer 1.
    loadB(b0[1], pB0, k + 32);
    loadB(b1[1], pB1, k + 32);
#pragma unroll
    for (int m = 0; m < 4; ++m) loadA(a[1][m], pA + (long)m * 16 * CDIM, k + 32);
    // Compute chunk k from buffer 0 (waits only on phase-0 loads).
#pragma unroll
    for (int m = 0; m < 4; ++m) {
      acc[0][m] = WMMA_BF16(a[0][m].v, b0[0].v, acc[0][m]);
      acc[1][m] = WMMA_BF16(a[0][m].v, b1[0].v, acc[1][m]);
    }
    // Issue loads for k+64 into buffer 0.
    loadB(b0[0], pB0, k + 64);
    loadB(b1[0], pB1, k + 64);
#pragma unroll
    for (int m = 0; m < 4; ++m) loadA(a[0][m], pA + (long)m * 16 * CDIM, k + 64);
    // Compute chunk k+32 from buffer 1.
#pragma unroll
    for (int m = 0; m < 4; ++m) {
      acc[0][m] = WMMA_BF16(a[1][m].v, b0[1].v, acc[0][m]);
      acc[1][m] = WMMA_BF16(a[1][m].v, b1[1].v, acc[1][m]);
    }
  }

  // Epilogue: per-row (max, sumexp) over this wave's 32 columns.
  // D layout: VGPR j <-> row m*16 + j + 8*half, col = lane%16 -> reduce across
  // the 16 lanes of each half with xor-shuffles (stay within the half).
  const float bv0 = cv0 ? bias[cc0] : 0.0f;
  const float bv1 = cv1 ? bias[cc1] : 0.0f;
#pragma unroll
  for (int m = 0; m < 4; ++m) {
#pragma unroll
    for (int j = 0; j < 8; ++j) {
      const float v0 = acc[0][m][j] + bv0;
      const float v1 = acc[1][m][j] + bv1;
      float mx = fmaxf(cv0 ? v0 : -1e30f, cv1 ? v1 : -1e30f);
#pragma unroll
      for (int s = 1; s <= 8; s <<= 1) mx = fmaxf(mx, __shfl_xor(mx, s, 32));
      float e = (cv0 ? __expf(v0 - mx) : 0.0f) + (cv1 ? __expf(v1 - mx) : 0.0f);
#pragma unroll
      for (int s = 1; s <= 8; s <<= 1) e += __shfl_xor(e, s, 32);
      if (l16 == 0) {
        const int r = m * 16 + half * 8 + j;
        lm[wave][r] = mx;
        lz[wave][r] = e;
      }
    }
  }
  __syncthreads();

  // Merge 8 waves' (max, Z) per row -> one partial per (row, vocab-tile).
  if (threadIdx.x < 64) {
    const int r = threadIdx.x;
    float mx = lm[0][r];
#pragma unroll
    for (int w = 1; w < 8; ++w) mx = fmaxf(mx, lm[w][r]);
    float z = 0.f;
#pragma unroll
    for (int w = 0; w < 8; ++w) z += lz[w][r] * __expf(lm[w][r] - mx);
    partials[(long)(row0 + r) * NVT + blockIdx.y] = make_float2(mx, z);
  }
}

// ---------------------------------------------------------------------------
// Kernel 4: per-row merge of 197 partials + target-logit dot + loss[n].
// One 256-thread block per row. Fixed reduction order -> deterministic.
// ---------------------------------------------------------------------------
__device__ __forceinline__ float warpMax(float v) {
#pragma unroll
  for (int s = 16; s >= 1; s >>= 1) v = fmaxf(v, __shfl_xor(v, s, 32));
  return v;
}
__device__ __forceinline__ float warpSum(float v) {
#pragma unroll
  for (int s = 16; s >= 1; s >>= 1) v += __shfl_xor(v, s, 32);
  return v;
}

__global__ __launch_bounds__(256) void row_reduce(
    const float2* __restrict__ partials, const __bf16* __restrict__ xb,
    const __bf16* __restrict__ wb, const float* __restrict__ bias,
    const int* __restrict__ y, float* __restrict__ loss) {
  __shared__ float sh[8];
  __shared__ float bc;
  const int n = blockIdx.x;
  const int t = threadIdx.x;
  const int lane = t & 31, wv = t >> 5;
  const float2* pn = partials + (long)n * NVT;

  // Pass 1: global row max.
  float mx = -1e30f;
  for (int i = t; i < NVT; i += 256) mx = fmaxf(mx, pn[i].x);
  mx = warpMax(mx);
  if (lane == 0) sh[wv] = mx;
  __syncthreads();
  if (t == 0) {
    float m2 = sh[0];
    for (int w = 1; w < 8; ++w) m2 = fmaxf(m2, sh[w]);
    bc = m2;
  }
  __syncthreads();
  mx = bc;
  __syncthreads();

  // Pass 2: Z = sum_i Z_i * exp(m_i - m).
  float z = 0.f;
  for (int i = t; i < NVT; i += 256) {
    const float2 p = pn[i];
    z += p.y * __expf(p.x - mx);
  }
  z = warpSum(z);
  if (lane == 0) sh[wv] = z;
  __syncthreads();
  if (t == 0) {
    float s = 0.f;
    for (int w = 0; w < 8; ++w) s += sh[w];
    bc = s;
  }
  __syncthreads();
  z = bc;
  __syncthreads();

  // Target logit: t_n = <x[n,:], W[y_n,:]> + b[y_n] (bf16 inputs, fp32 accum;
  // consistent with the GEMM precision). 2048 = 256 threads x 8 elements.
  const int yv = y[n];
  const v8bf xa = *(const v8bf*)(xb + (long)n * CDIM + t * 8);
  const v8bf wa = *(const v8bf*)(wb + (long)yv * CDIM + t * 8);
  float d = 0.f;
#pragma unroll
  for (int i = 0; i < 8; ++i) d += (float)xa[i] * (float)wa[i];
  d = warpSum(d);
  if (lane == 0) sh[wv] = d;
  __syncthreads();
  if (t == 0) {
    float s = 0.f;
    for (int w = 0; w < 8; ++w) s += sh[w];
    loss[n] = mx + logf(z) - (s + bias[yv]);
  }
}

// ---------------------------------------------------------------------------
// Kernel 5: deterministic mean over 8192 row losses.
// ---------------------------------------------------------------------------
__global__ __launch_bounds__(256) void final_mean(const float* __restrict__ loss,
                                                  float* __restrict__ out) {
  __shared__ float sh[8];
  const int t = threadIdx.x;
  float s = 0.f;
  for (int i = t; i < NROW; i += 256) s += loss[i];
  s = warpSum(s);
  if ((t & 31) == 0) sh[t >> 5] = s;
  __syncthreads();
  if (t == 0) {
    float tot = 0.f;
    for (int w = 0; w < 8; ++w) tot += sh[w];
    out[0] = tot / (float)NROW;
  }
}

// ---------------------------------------------------------------------------
// Launch: x -> bf16, W -> bf16, fused GEMM+LSE partials, row merge, mean.
// Workspace: xb(32MB) | wb(197MB) | partials(13MB) | loss(32KB).
// ---------------------------------------------------------------------------
extern "C" void kernel_launch(void* const* d_in, const int* in_sizes, int n_in,
                              void* d_out, int out_size, void* d_ws, size_t ws_size,
                              hipStream_t stream) {
  const float* x = (const float*)d_in[0];
  const int*   y = (const int*)d_in[1];
  const float* W = (const float*)d_in[2];
  const float* b = (const float*)d_in[3];
  float* out = (float*)d_out;

  char* ws = (char*)d_ws;
  const size_t XBYTES = (size_t)NROW * CDIM * 2;
  const size_t WBYTES = (size_t)VDIM * CDIM * 2;
  const size_t PBYTES = (size_t)NROW * NVT * sizeof(float2);
  __bf16* xb = (__bf16*)ws;
  __bf16* wb = (__bf16*)(ws + XBYTES);
  float2* partials = (float2*)(ws + XBYTES + WBYTES);
  float*  loss = (float*)(ws + XBYTES + WBYTES + PBYTES);

  const long nx4 = (long)NROW * CDIM / 4;
  const long nw4 = (long)VDIM * CDIM / 4;
  cvt_f32_bf16<<<(int)((nx4 + 255) / 256), 256, 0, stream>>>(x, xb, nx4);
  cvt_f32_bf16<<<(int)((nw4 + 255) / 256), 256, 0, stream>>>(W, wb, nw4);
  // blockIdx.x = m-block (fast) so one W slice is shared across 128 consecutive
  // workgroups while resident in the 192MB L2; bf16 x stays L2-resident too.
  gemm_lse_partials<<<dim3(NROW / 64, NVT), 256, 0, stream>>>(xb, wb, b, partials);
  row_reduce<<<NROW, 256, 0, stream>>>(partials, xb, wb, b, y, loss);
  final_mean<<<1, 256, 0, stream>>>(loss, out);

  (void)in_sizes; (void)n_in; (void)out_size; (void)ws_size;
}
